// SSMLayer_48172353192498
// MI455X (gfx1250) — matrix operand order repaired and checked by reading, not compiled, not run on hardware
//
#include <hip/hip_runtime.h>

typedef __attribute__((ext_vector_type(16))) __bf16 v16bf;
typedef __attribute__((ext_vector_type(8)))  float  v8f;

#define SSM_N   256
#define SSM_L   4096
#define SSM_B   1024
#define MATF    (SSM_N * SSM_N)      // 65536 floats per 256x256 matrix
#define NKREV   4160                 // reversed+padded K taps (bf16 bits)
#define NTERMS  16                   // Taylor terms for expm / phi1

// ---- workspace layout (float offsets) ----
static constexpr size_t OFF_AS    = 0;                   // X = step*A
static constexpr size_t OFF_TA    = OFF_AS    + MATF;    // Taylor ping / squaring tmp
static constexpr size_t OFF_TB    = OFF_TA    + MATF;    // Taylor pong / squaring tmp
static constexpr size_t OFF_AB    = OFF_TB    + MATF;    // Ab = expm(step*A)
static constexpr size_t OFF_M16   = OFF_AB    + MATF;    // Ab^16
static constexpr size_t OFF_M256  = OFF_M16   + MATF;    // Ab^256
static constexpr size_t OFF_LROWS = OFF_M256  + MATF;    // 256 x 256 left rows
static constexpr size_t OFF_BB    = OFF_LROWS + MATF;    // Bb (256)
static constexpr size_t OFF_R     = OFF_BB    + 256;     // R 256x16 row-major
static constexpr size_t OFF_K     = OFF_R     + 4096;    // K taps f32 (4096)
static constexpr size_t OFF_KREV  = OFF_K     + 4096;    // NKREV ushorts (bf16 bits)
static constexpr size_t OFF_UBF   = OFF_KREV  + NKREV / 2; // u in bf16 bits (B*L ushorts)

__device__ __forceinline__ unsigned short bfbits(float f) {
    unsigned u = __float_as_uint(f);
    return (unsigned short)((u + 0x7FFFu + ((u >> 16) & 1u)) >> 16);   // RNE f32->bf16
}

// ---------- setup: X = step*A ; Acc(Ab) = I + X ----------
__global__ __launch_bounds__(256) void ssm_scale(const float* __restrict__ A,
                                                 const float* __restrict__ ls,
                                                 float* __restrict__ X,
                                                 float* __restrict__ Acc) {
    int i = blockIdx.x, j = threadIdx.x;
    float step = __expf(ls[0]);
    float v = step * A[i * SSM_N + j];
    X[i * SSM_N + j] = v;
    Acc[i * SSM_N + j] = v + (i == j ? 1.0f : 0.0f);
}

// ---------- Taylor term: Tnew = Told * X * invk ; Acc += Tnew ----------
__global__ __launch_bounds__(256) void ssm_term(const float* __restrict__ Told,
                                                const float* __restrict__ X,
                                                float* __restrict__ Tnew,
                                                float* __restrict__ Acc, float invk) {
    __shared__ float row[SSM_N];
    int i = blockIdx.x, j = threadIdx.x;
    row[j] = Told[i * SSM_N + j];
    __syncthreads();
    float s = 0.f;
    for (int k = 0; k < SSM_N; ++k) s = fmaf(row[k], X[k * SSM_N + j], s);
    s *= invk;
    Tnew[i * SSM_N + j] = s;
    Acc[i * SSM_N + j] += s;
}

// ---------- generic C = A*B (256x256) ----------
__global__ __launch_bounds__(256) void ssm_mm(const float* __restrict__ A,
                                              const float* __restrict__ B,
                                              float* __restrict__ C) {
    __shared__ float row[SSM_N];
    int i = blockIdx.x, j = threadIdx.x;
    row[j] = A[i * SSM_N + j];
    __syncthreads();
    float s = 0.f;
    for (int k = 0; k < SSM_N; ++k) s = fmaf(row[k], B[k * SSM_N + j], s);
    C[i * SSM_N + j] = s;
}

// ---------- Bb = step * phi1(X) * B  (vector Taylor, no solve) ----------
__global__ __launch_bounds__(256) void ssm_bb(const float* __restrict__ X,
                                              const float* __restrict__ B,
                                              const float* __restrict__ ls,
                                              float* __restrict__ Bb) {
    __shared__ float sv[SSM_N];
    int i = threadIdx.x;
    float acc = B[i];          // k=0 term, 1/1!
    sv[i] = B[i];
    __syncthreads();
    float fact = 1.0f;
    for (int k = 1; k <= NTERMS; ++k) {
        float s = 0.f;
        for (int j = 0; j < SSM_N; ++j) s = fmaf(X[i * SSM_N + j], sv[j], s);
        fact *= (float)(k + 1);
        acc += s / fact;       // X^k B / (k+1)!
        __syncthreads();
        sv[i] = s;
        __syncthreads();
    }
    Bb[i] = __expf(ls[0]) * acc;
}

// ---------- left row scan over Ab^256: rows a*16 of Lrows ----------
__global__ __launch_bounds__(256) void ssm_lscan_a(const float* __restrict__ Cm,
                                                   const float* __restrict__ M256,
                                                   float* __restrict__ Lrows) {
    __shared__ float l[SSM_N];
    int j = threadIdx.x;
    l[j] = Cm[j];
    Lrows[j] = l[j];
    __syncthreads();
    for (int a = 1; a < 16; ++a) {
        float s = 0.f;
        for (int k = 0; k < SSM_N; ++k) s = fmaf(l[k], M256[k * SSM_N + j], s);
        __syncthreads();
        l[j] = s;
        __syncthreads();
        Lrows[(size_t)(a * 16) * SSM_N + j] = s;
    }
}

// ---------- left row scan over Ab^16: fills rows a*16+b ----------
__global__ __launch_bounds__(256) void ssm_lscan_b(const float* __restrict__ M16,
                                                   float* __restrict__ Lrows) {
    __shared__ float l[SSM_N];
    int a = blockIdx.x, j = threadIdx.x;
    l[j] = Lrows[(size_t)(a * 16) * SSM_N + j];
    __syncthreads();
    for (int b = 1; b < 16; ++b) {
        float s = 0.f;
        for (int k = 0; k < SSM_N; ++k) s = fmaf(l[k], M16[k * SSM_N + j], s);
        __syncthreads();
        l[j] = s;
        __syncthreads();
        Lrows[(size_t)(a * 16 + b) * SSM_N + j] = s;
    }
}

// ---------- right scan: R[:,c] = Ab^c * Bb  (row-major 256x16) ----------
__global__ __launch_bounds__(256) void ssm_rscan(const float* __restrict__ Ab,
                                                 const float* __restrict__ Bb,
                                                 float* __restrict__ R) {
    __shared__ float sv[SSM_N];
    int i = threadIdx.x;
    sv[i] = Bb[i];
    R[i * 16 + 0] = Bb[i];
    __syncthreads();
    for (int c = 1; c < 16; ++c) {
        float s = 0.f;
        for (int k = 0; k < SSM_N; ++k) s = fmaf(Ab[i * SSM_N + k], sv[k], s);
        __syncthreads();
        sv[i] = s;
        __syncthreads();
        R[i * 16 + c] = s;
    }
}

// ---------- K = Lrows(256x256) * R(256x16), flattened row-major = K[l] ----------
__global__ __launch_bounds__(256) void ssm_kmat(const float* __restrict__ Lrows,
                                                const float* __restrict__ R,
                                                float* __restrict__ K) {
    int idx = blockIdx.x * 256 + threadIdx.x;   // l = r*16 + c
    int r = idx >> 4, c = idx & 15;
    float s = 0.f;
    for (int k = 0; k < SSM_N; ++k) s = fmaf(Lrows[(size_t)r * SSM_N + k], R[k * 16 + c], s);
    K[idx] = s;
}

// ---------- reversed, zero-padded bf16 taps: Krev[i] = K[4096-i] ----------
__global__ __launch_bounds__(256) void ssm_krev(const float* __restrict__ K,
                                                unsigned short* __restrict__ Krev) {
    int i = blockIdx.x * 256 + threadIdx.x;
    if (i >= NKREV) return;
    int src = SSM_L - i;
    Krev[i] = (src >= 0 && src < SSM_L) ? bfbits(K[src]) : (unsigned short)0;
}

// ---------- one-time u f32 -> bf16 copy (stays L2-resident for the conv) ----------
__global__ __launch_bounds__(256) void ssm_ubf(const float* __restrict__ u,
                                               unsigned short* __restrict__ ub) {
    size_t i = ((size_t)blockIdx.x * 256 + threadIdx.x) * 8;
    const float4* p = (const float4*)(u + i);
    float4 a = p[0], b = p[1];
    unsigned short o[8] = {bfbits(a.x), bfbits(a.y), bfbits(a.z), bfbits(a.w),
                           bfbits(b.x), bfbits(b.y), bfbits(b.z), bfbits(b.w)};
    __builtin_memcpy(ub + i, o, 16);
}

// ---------- main causal Toeplitz conv: y = u (*) K + D*u, via WMMA bf16 ----------
// grid (L/64, BATCH/128), 256 threads = 8 waves. Wave w: 16 batch rows x 64
// time cols (4 accumulators); one A fragment feeds 4 WMMAs per k-step.
// B fragments are 4 shifted 32B windows of the LDS-staged reversed taps
// (contiguous per lane thanks to the reversal; zero padding = causal mask).
__global__ __launch_bounds__(256) void ssm_conv(const unsigned short* __restrict__ ubf,
                                                const float* __restrict__ u,
                                                const unsigned short* __restrict__ krev,
                                                const float* __restrict__ Dp,
                                                float* __restrict__ y) {
    __shared__ unsigned short sk[NKREV];
    for (int i = threadIdx.x; i < NKREV; i += 256) sk[i] = krev[i];
    __syncthreads();

    const int lane = threadIdx.x & 31;
    const int wave = threadIdx.x >> 5;
    const int half = lane >> 4;            // 0 / 1
    const int nl   = lane & 15;
    const int m    = blockIdx.y * 128 + wave * 16 + nl;   // A-row for this lane
    const int kbA  = half * 16;            // A: lanes 0-15 hold K 0..15
    const int kbB  = half * 16;            // B: lanes 0-15 hold K 0..15
    const int t0   = blockIdx.x * 64;
    const int ta   = t0 + nl;
    const unsigned short* __restrict__ urow = ubf + (size_t)m * SSM_L;

    v8f acc0 = {}, acc1 = {}, acc2 = {}, acc3 = {};

    const int send = t0 + 32;              // last s0 (multiple of 32 covering t0+63)
    for (int s0 = 0; s0 <= send; s0 += 32) {
        // ---- A fragment: two aligned 16B bf16 loads (K {0..7} and {8..15} + half*16)
        unsigned short ab[16];
        __builtin_memcpy(ab,     urow + s0 + kbA,      16);
        __builtin_memcpy(ab + 8, urow + s0 + kbA + 8,  16);
        v16bf av;
        __builtin_memcpy(&av, ab, 32);

        // ---- B fragments: W[s,t]=K[t-s]; four shifted windows of reversed taps
        int base0 = SSM_L - ta + s0 + kbB;
        v16bf b0, b1, b2, b3;
        __builtin_memcpy(&b0, &sk[base0],      32);
        __builtin_memcpy(&b1, &sk[base0 - 16], 32);
        __builtin_memcpy(&b2, &sk[base0 - 32], 32);
        __builtin_memcpy(&b3, &sk[base0 - 48], 32);

        acc0 = __builtin_amdgcn_wmma_f32_16x16x32_bf16(false, av, false, b0,
                                                       (short)0, acc0, false, false);
        acc1 = __builtin_amdgcn_wmma_f32_16x16x32_bf16(false, av, false, b1,
                                                       (short)0, acc1, false, false);
        acc2 = __builtin_amdgcn_wmma_f32_16x16x32_bf16(false, av, false, b2,
                                                       (short)0, acc2, false, false);
        acc3 = __builtin_amdgcn_wmma_f32_16x16x32_bf16(false, av, false, b3,
                                                       (short)0, acc3, false, false);
    }

    // ---- epilogue: f32 D*u skip, scatter per C/D layout (VGPR r -> M = r + 8*half)
    const float D = Dp[0];
    const int mr0 = blockIdx.y * 128 + wave * 16 + half * 8;
#pragma unroll
    for (int r = 0; r < 8; ++r) {
        size_t ro = (size_t)(mr0 + r) * SSM_L;
        int t = ta;
        y[ro + t] = acc0[r] + D * u[ro + t];
        t += 16;
        y[ro + t] = acc1[r] + D * u[ro + t];
        t += 16;
        y[ro + t] = acc2[r] + D * u[ro + t];
        t += 16;
        y[ro + t] = acc3[r] + D * u[ro + t];
    }
}

extern "C" void kernel_launch(void* const* d_in, const int* in_sizes, int n_in,
                              void* d_out, int out_size, void* d_ws, size_t ws_size,
                              hipStream_t stream) {
    (void)in_sizes; (void)n_in; (void)out_size; (void)ws_size;
    const float* u  = (const float*)d_in[0];
    const float* A  = (const float*)d_in[1];
    const float* B  = (const float*)d_in[2];
    const float* Cm = (const float*)d_in[3];
    const float* D  = (const float*)d_in[4];
    const float* ls = (const float*)d_in[5];

    float* ws    = (float*)d_ws;
    float* As    = ws + OFF_AS;
    float* Ta    = ws + OFF_TA;
    float* Tb    = ws + OFF_TB;
    float* Ab    = ws + OFF_AB;
    float* M16   = ws + OFF_M16;
    float* M256  = ws + OFF_M256;
    float* Lrows = ws + OFF_LROWS;
    float* Bb    = ws + OFF_BB;
    float* R     = ws + OFF_R;
    float* Kf    = ws + OFF_K;
    unsigned short* Krev = (unsigned short*)(ws + OFF_KREV);
    unsigned short* Ubf  = (unsigned short*)(ws + OFF_UBF);

    // 0) one-time u -> bf16 copy (independent of the K pipeline)
    ssm_ubf<<<(SSM_B * SSM_L) / (256 * 8), 256, 0, stream>>>(u, Ubf);

    // 1) X = step*A ; Ab = I + X (Taylor accumulator)
    ssm_scale<<<SSM_N, 256, 0, stream>>>(A, ls, As, Ab);

    // 2) expm(step*A) via Taylor: T_k = T_{k-1}*X/k, Ab += T_k, k=2..16
    const float* told = As;
    float* tnew = Ta;
    for (int k = 2; k <= NTERMS; ++k) {
        ssm_term<<<SSM_N, 256, 0, stream>>>(told, As, tnew, Ab, 1.0f / (float)k);
        told = tnew;
        tnew = (tnew == Ta) ? Tb : Ta;
    }

    // 3) Bb = step * phi1(step*A) * B
    ssm_bb<<<1, 256, 0, stream>>>(As, B, ls, Bb);

    // 4) squarings: M16 = Ab^16, M256 = Ab^256  (Ta/Tb reused as temps)
    ssm_mm<<<SSM_N, 256, 0, stream>>>(Ab, Ab, Ta);     // ^2
    ssm_mm<<<SSM_N, 256, 0, stream>>>(Ta, Ta, Tb);     // ^4
    ssm_mm<<<SSM_N, 256, 0, stream>>>(Tb, Tb, Ta);     // ^8
    ssm_mm<<<SSM_N, 256, 0, stream>>>(Ta, Ta, M16);    // ^16
    ssm_mm<<<SSM_N, 256, 0, stream>>>(M16, M16, Ta);   // ^32
    ssm_mm<<<SSM_N, 256, 0, stream>>>(Ta, Ta, Tb);     // ^64
    ssm_mm<<<SSM_N, 256, 0, stream>>>(Tb, Tb, Ta);     // ^128
    ssm_mm<<<SSM_N, 256, 0, stream>>>(Ta, Ta, M256);   // ^256

    // 5) K[l] = C Ab^l Bb via l = 256a + 16b + c decomposition
    ssm_lscan_a<<<1, 256, 0, stream>>>(Cm, M256, Lrows);
    ssm_lscan_b<<<16, 256, 0, stream>>>(M16, Lrows);
    ssm_rscan<<<1, 256, 0, stream>>>(Ab, Bb, R);
    ssm_kmat<<<16, 256, 0, stream>>>(Lrows, R, Kf);
    ssm_krev<<<(NKREV + 255) / 256, 256, 0, stream>>>(Kf, Krev);

    // 6) causal Toeplitz conv via bf16 WMMA, D*u fused in f32 epilogue
    dim3 grid(SSM_L / 64, SSM_B / 128);
    ssm_conv<<<grid, 256, 0, stream>>>(Ubf, u, Krev, D, (float*)d_out);
}
